// DynamicSNN_38637525795316
// MI455X (gfx1250) — compile-verified, matrix operand order checked by reading, not run on hardware
//
#include <hip/hip_runtime.h>

// ---------------- problem dims ----------------
#define T_STEPS 1000
#define BATCH   64
#define DIN     700
#define DH1     400
#define DH2     400
#define DOUT    20

// K padded to multiples of 32 (WMMA bf16 K-tile)
#define KP1     704
#define KP2     416
#define KT1     (KP1 / 32)   // 22
#define KT2     (KP2 / 32)   // 13
#define NWG     25           // 25 * 16 = 400 output columns per hidden layer

// LDS pitches
#define LPA     712          // layer-1 activation stage pitch (>= 704)
#define LPW1    712          // W1 tile pitch
#define LPW2    424          // W2/W3 tile pitch (>= 416)
// layers 2/3 reuse shA with pitch KP2 (=416), matching the TDM tile layout

typedef __attribute__((ext_vector_type(16))) __bf16 v16bf;
typedef __attribute__((ext_vector_type(8)))  __bf16 v8bf;
typedef __attribute__((ext_vector_type(4)))  __bf16 v4bf;
typedef __attribute__((ext_vector_type(8)))  float  v8f;
typedef __attribute__((ext_vector_type(4)))  unsigned int v4u;
typedef __attribute__((ext_vector_type(8)))  int    v8i;
typedef __attribute__((ext_vector_type(4)))  int    v4i;

#if defined(__has_builtin)
#if __has_builtin(__builtin_amdgcn_tensor_load_to_lds) && \
    __has_builtin(__builtin_amdgcn_s_wait_tensorcnt)
#define USE_TDM 1
#endif
#endif
#ifndef USE_TDM
#define USE_TDM 0
#endif

// ---- WMMA fragment load: 16-bit A/B layout (ISA 7.12.2) ----
// lane L: row = L&15, chunk0 at k0=(L>>4)*8, chunk1 at k0+16 (both 16B, aligned)
__device__ __forceinline__ v16bf ld_frag(const __bf16* p) {
  union { v16bf v; v8bf h[2]; } u;
  u.h[0] = *(const v8bf*)(p);
  u.h[1] = *(const v8bf*)(p + 16);
  return u.v;
}

// C = A(16xK) * B(KxN=16) with bf16 inputs, f32 accum
__device__ __forceinline__ v8f gemm_tile(const __bf16* A, int apitch,
                                         const __bf16* Wt, int wpitch,
                                         int lane, int nkt) {
  v8f acc = {0.f, 0.f, 0.f, 0.f, 0.f, 0.f, 0.f, 0.f};
  const int row  = lane & 15;
  const int koff = (lane >> 4) << 3;
  const __bf16* pa = A  + row * apitch + koff;
  const __bf16* pb = Wt + row * wpitch + koff;
  for (int kt = 0; kt < nkt; ++kt) {
    v16bf a = ld_frag(pa + kt * 32);
    v16bf b = ld_frag(pb + kt * 32);
    acc = __builtin_amdgcn_wmma_f32_16x16x32_bf16(false, a, false, b,
                                                  (short)0, acc, false, false);
  }
  return acc;
}

#if USE_TDM
// TDM: DMA a 2D bf16 tile (tile_d0 x tile_d1) from a row-major global tensor
// (tensor_d0 x tensor_d1, row stride stride0 elements) into LDS at lds_off.
// Reads past tensor_dim0 return zero -> free K padding. D# per ISA §8.
// This toolchain's builtin is the 6-arg form:
//   (uint32x4 g0, int32x8 g1, int32x4 g2, int32x4 g3, int32x8 extra, i32 cpol)
__device__ __forceinline__ void tdm_load_2d(unsigned lds_off, const void* gaddr,
                                            unsigned tensor_d0, unsigned tensor_d1,
                                            unsigned tile_d0, unsigned tile_d1,
                                            unsigned stride0) {
  unsigned long long ga = (unsigned long long)(uintptr_t)gaddr;
  v4u g0;
  g0.x = 1u;                                            // count=1 (valid), user mode
  g0.y = lds_off;                                       // LDS byte address
  g0.z = (unsigned)(ga & 0xFFFFFFFFull);                // global_addr[31:0]
  g0.w = (unsigned)((ga >> 32) & 0x1FFFFFFull)          // global_addr[56:32]
       | (2u << 30);                                    // type = 2 ("image")
  v8i g1;
  g1[0] = (int)(1u << 16);                              // data_size=1 -> 2 bytes
  g1[1] = (int)((tensor_d0 & 0xFFFFu) << 16);           // tensor_dim0[15:0]
  g1[2] = (int)((tensor_d0 >> 16) | ((tensor_d1 & 0xFFFFu) << 16));
  g1[3] = (int)((tensor_d1 >> 16) | (tile_d0 << 16));   // tile_dim0
  g1[4] = (int)(tile_d1 & 0xFFFFu);                     // tile_dim1 (tile_dim2=0)
  g1[5] = (int)stride0;                                 // tensor_dim0_stride lo32
  g1[6] = 0;                                            // stride0 hi / stride1 lo
  g1[7] = 0;
  v4i g2 = {0, 0, 0, 0};                                // dims 2/3: unused (2D)
  v4i g3 = {0, 0, 0, 0};
  v8i g4 = {0, 0, 0, 0, 0, 0, 0, 0};                    // extra group: unused
  __builtin_amdgcn_tensor_load_to_lds(g0, g1, g2, g3, g4, 0);
}
#endif

// device-wide generation-counter barrier (bar[0]=count, bar[1]=generation)
__device__ __forceinline__ void grid_barrier(unsigned* bar) {
  __syncthreads();
  if (threadIdx.x == 0) {
    __threadfence();
    unsigned* cnt = bar;
    unsigned* gen = bar + 1;
    unsigned g = __hip_atomic_load(gen, __ATOMIC_RELAXED, __HIP_MEMORY_SCOPE_AGENT);
    unsigned a = __hip_atomic_fetch_add(cnt, 1u, __ATOMIC_ACQ_REL, __HIP_MEMORY_SCOPE_AGENT);
    if (a == NWG - 1u) {
      __hip_atomic_store(cnt, 0u, __ATOMIC_RELAXED, __HIP_MEMORY_SCOPE_AGENT);
      __hip_atomic_fetch_add(gen, 1u, __ATOMIC_RELEASE, __HIP_MEMORY_SCOPE_AGENT);
    } else {
      while (__hip_atomic_load(gen, __ATOMIC_ACQUIRE, __HIP_MEMORY_SCOPE_AGENT) == g)
        __builtin_amdgcn_s_sleep(1);
    }
    __threadfence();
  }
  __syncthreads();
}

__global__ void init_bar_kernel(unsigned* bar) {
  if (threadIdx.x == 0) { bar[0] = 0u; bar[1] = 0u; }
}

__global__ __launch_bounds__(128, 1)
void snn_persistent(const float* __restrict__ s,
                    const float* __restrict__ W1,
                    const float* __restrict__ W2,
                    const float* __restrict__ W3,
                    const float* __restrict__ tw1,
                    const float* __restrict__ tw2,
                    const float* __restrict__ tw3,
                    float* __restrict__ out_s,
                    float* __restrict__ out_v,
                    __bf16* __restrict__ spk1,
                    __bf16* __restrict__ spk2,
                    unsigned* __restrict__ bar) {
  __shared__ __bf16 shA [BATCH * LPA];   // layer1: [64][712]; layers2/3: [64][416]
  __shared__ __bf16 shW1[16 * LPW1];
  __shared__ __bf16 shW2[16 * LPW2];
  __shared__ __bf16 shW3[16 * LPW2];

  const int tid   = threadIdx.x;
  const int wg    = blockIdx.x;          // 0..24 : owns N cols [wg*16, wg*16+16)
  const int lane  = tid & 31;
  const int wave  = tid >> 5;            // 0..3 : owns M rows [wave*16, wave*16+16)
  const int ncol  = lane & 15;
  const int mhalf = (lane >> 4) << 3;
  const int n12   = wg * 16 + ncol;

  // ---------- one-time weight preload (persistent in LDS) ----------
  for (int idx = tid; idx < 16 * KP1; idx += 128) {
    int r = idx / KP1, k = idx - r * KP1;
    float w = (k < DIN) ? W1[(wg * 16 + r) * DIN + k] : 0.f;
    shW1[r * LPW1 + k] = (__bf16)w;
  }
  for (int idx = tid; idx < 16 * KP2; idx += 128) {
    int r = idx / KP2, k = idx - r * KP2;
    float w = (k < DH1) ? W2[(wg * 16 + r) * DH1 + k] : 0.f;
    shW2[r * LPW2 + k] = (__bf16)w;
  }
  if (wg < 2) {
    for (int idx = tid; idx < 16 * KP2; idx += 128) {
      int r = idx / KP2, k = idx - r * KP2;
      int n = wg * 16 + r;
      float w = (n < DOUT && k < DH2) ? W3[n * DH2 + k] : 0.f;
      shW3[r * LPW2 + k] = (__bf16)w;
    }
  }

  const float dt1 = 0.01f / (0.1f + 1.f / (1.f + __expf(-tw1[n12])));
  const float dt2 = 0.01f / (0.1f + 1.f / (1.f + __expf(-tw2[n12])));
  float dt3 = 0.f;
  if (wg < 2 && n12 < DOUT)
    dt3 = 0.01f / (0.1f + 1.f / (1.f + __expf(-tw3[n12])));

  v8f v1 = {0.f,0.f,0.f,0.f,0.f,0.f,0.f,0.f};
  v8f v2 = {0.f,0.f,0.f,0.f,0.f,0.f,0.f,0.f};
  v8f v3 = {0.f,0.f,0.f,0.f,0.f,0.f,0.f,0.f};

  for (int t = 0; t < T_STEPS; ++t) {
    __syncthreads();                        // shA free (prev-iter readers done)

    // ---- stage input spikes s[t] (f32 -> bf16), pitch LPA, pad K to 704 ----
    const float* st = s + (size_t)t * (BATCH * DIN);
    for (int idx = tid; idx < BATCH * (KP1 / 4); idx += 128) {
      int m  = idx / (KP1 / 4);
      int k4 = idx - m * (KP1 / 4);
      float4 val = {0.f, 0.f, 0.f, 0.f};
      if (k4 < DIN / 4) val = *(const float4*)(st + m * DIN + k4 * 4);
      __bf16* dst = &shA[m * LPA + k4 * 4];
      dst[0] = (__bf16)val.x; dst[1] = (__bf16)val.y;
      dst[2] = (__bf16)val.z; dst[3] = (__bf16)val.w;
    }
    if (t + 1 < T_STEPS)
      __builtin_prefetch((const char*)(st + BATCH * DIN) + tid * 1408, 0, 1);
    __syncthreads();

    // ---- layer 1 ----
    {
      v8f cur = gemm_tile(shA + wave * 16 * LPA, LPA, shW1, LPW1, lane, KT1);
#pragma unroll
      for (int r = 0; r < 8; ++r) {
        float vn = v1[r] + dt1 * (cur[r] - v1[r]);   // v + dt*((0-v)+cur)
        float sp = (vn > 1.0f) ? 1.0f : 0.0f;        // heaviside(v_new - vth)
        v1[r] = vn - vn * sp;                        // reset-to-zero
        spk1[(wave * 16 + mhalf + r) * DH1 + n12] = (__bf16)sp;
      }
    }
    grid_barrier(bar);

    // ---- stage spk1 -> shA as [64][416] (TDM async copy, OOB -> zero pad) ----
#if USE_TDM
    if (wave == 0) {
      tdm_load_2d((unsigned)(uintptr_t)shA, spk1, DH1, BATCH, KP2, BATCH, DH1);
      __builtin_amdgcn_s_wait_tensorcnt(0);
    }
#else
    for (int idx = tid; idx < BATCH * (KP2 / 4); idx += 128) {
      int m  = idx / (KP2 / 4);
      int k4 = idx - m * (KP2 / 4);
      v4bf val = {(__bf16)0.f, (__bf16)0.f, (__bf16)0.f, (__bf16)0.f};
      if (k4 < DH1 / 4) val = *(const v4bf*)(spk1 + m * DH1 + k4 * 4);
      *(v4bf*)(&shA[m * KP2 + k4 * 4]) = val;
    }
#endif
    __syncthreads();

    // ---- layer 2 ----
    {
      v8f cur = gemm_tile(shA + wave * 16 * KP2, KP2, shW2, LPW2, lane, KT2);
#pragma unroll
      for (int r = 0; r < 8; ++r) {
        float vn = v2[r] + dt2 * (cur[r] - v2[r]);
        float sp = (vn > 1.0f) ? 1.0f : 0.0f;
        v2[r] = vn - vn * sp;
        spk2[(wave * 16 + mhalf + r) * DH2 + n12] = (__bf16)sp;
      }
    }
    grid_barrier(bar);

    // ---- layer 3 (OUT=20 -> WGs 0 and 1 only; branch is WG-uniform) ----
    if (wg < 2) {
#if USE_TDM
      if (wave == 0) {
        tdm_load_2d((unsigned)(uintptr_t)shA, spk2, DH2, BATCH, KP2, BATCH, DH2);
        __builtin_amdgcn_s_wait_tensorcnt(0);
      }
#else
      for (int idx = tid; idx < BATCH * (KP2 / 4); idx += 128) {
        int m  = idx / (KP2 / 4);
        int k4 = idx - m * (KP2 / 4);
        v4bf val = {(__bf16)0.f, (__bf16)0.f, (__bf16)0.f, (__bf16)0.f};
        if (k4 < DH2 / 4) val = *(const v4bf*)(spk2 + m * DH2 + k4 * 4);
        *(v4bf*)(&shA[m * KP2 + k4 * 4]) = val;
      }
#endif
      __syncthreads();
      v8f cur = gemm_tile(shA + wave * 16 * KP2, KP2, shW3, LPW2, lane, KT2);
#pragma unroll
      for (int r = 0; r < 8; ++r) {
        float vn = v3[r] + dt3 * (cur[r] - v3[r]);
        float sp = (vn > 1.0f) ? 1.0f : 0.0f;
        v3[r] = vn - vn * sp;
        if (n12 < DOUT) {
          size_t o = (size_t)t * (BATCH * DOUT)
                   + (size_t)(wave * 16 + mhalf + r) * DOUT + n12;
          out_s[o] = sp;
          out_v[o] = vn;
        }
      }
    }
  }
}

extern "C" void kernel_launch(void* const* d_in, const int* in_sizes, int n_in,
                              void* d_out, int out_size, void* d_ws, size_t ws_size,
                              hipStream_t stream) {
  (void)in_sizes; (void)n_in; (void)out_size; (void)ws_size;
  const float* s   = (const float*)d_in[0];
  const float* W1  = (const float*)d_in[1];
  const float* W2  = (const float*)d_in[2];
  const float* W3  = (const float*)d_in[3];
  const float* tw1 = (const float*)d_in[4];
  const float* tw2 = (const float*)d_in[5];
  const float* tw3 = (const float*)d_in[6];

  float* out_s = (float*)d_out;                          // [T,B,OUT]
  float* out_v = out_s + (size_t)T_STEPS * BATCH * DOUT; // [T,B,OUT]

  unsigned* bar  = (unsigned*)d_ws;
  __bf16*   spk1 = (__bf16*)((char*)d_ws + 256);
  __bf16*   spk2 = (__bf16*)((char*)d_ws + 256 + (size_t)BATCH * DH1 * 2);

  init_bar_kernel<<<1, 32, 0, stream>>>(bar);
  snn_persistent<<<dim3(NWG), dim3(128), 0, stream>>>(
      s, W1, W2, W3, tw1, tw2, tw3, out_s, out_v, spk1, spk2, bar);
}
